// RWKV6Attention_3487513444480
// MI455X (gfx1250) — matrix operand order repaired and checked by reading, not compiled, not run on hardware
//
#include <hip/hip_runtime.h>
#include <hip/hip_bf16.h>

// ---------------- problem constants ----------------
#define Bb 8
#define Tt 512
#define Cc 2048
#define Hh 32
#define Zz 64
#define Aa 2048
#define Mm (Bb * Tt) /* 4096 */
#define TMX 32
#define TDX 64
#define GN_EPS (1e-5f * 64.0f)

typedef __attribute__((ext_vector_type(16))) __bf16 v16bf;
typedef __attribute__((ext_vector_type(8))) float v8f;

__device__ __forceinline__ unsigned short f2bf(float f) {
  unsigned u = __float_as_uint(f);
  u += 0x7FFFu + ((u >> 16) & 1u); // round-to-nearest-even
  return (unsigned short)(u >> 16);
}

// ---------------- pack kernels ----------------
__global__ void cvt_bf16_kernel(const float* __restrict__ in,
                                unsigned short* __restrict__ out, int n) {
  int i = blockIdx.x * 256 + threadIdx.x;
  if (i < n) out[i] = f2bf(in[i]);
}

// in [K,N] f32 row-major -> out [N,K] bf16 row-major (pre-transposed weights)
__global__ void cvt_bf16_t_kernel(const float* __restrict__ in,
                                  unsigned short* __restrict__ out, int K,
                                  int N) {
  size_t i = (size_t)blockIdx.x * 256 + threadIdx.x;
  if (i < (size_t)K * N) {
    int k = (int)(i / N), n = (int)(i % N);
    out[(size_t)n * K + k] = f2bf(in[i]);
  }
}

// w2 [5,32,C] -> w2t [5,C,32]
__global__ void w2_transpose_kernel(const float* __restrict__ w2,
                                    float* __restrict__ w2t) {
  int i = blockIdx.x * 256 + threadIdx.x;
  int n = 5 * TMX * Cc;
  if (i < n) {
    int f = i / (TMX * Cc);
    int rem = i - f * (TMX * Cc);
    int e = rem / Cc;
    int c = rem - e * Cc;
    w2t[((size_t)f * Cc + c) * TMX + e] = w2[i];
  }
}

// xxx = x + (shift_eff - x) * maa_x  -> bf16
__global__ void prep_xxx_kernel(const float* __restrict__ x,
                                const int* __restrict__ pos,
                                const float* __restrict__ shift,
                                const float* __restrict__ maa_x,
                                unsigned short* __restrict__ xxxb) {
  size_t i = (size_t)blockIdx.x * 256 + threadIdx.x;
  if (i >= (size_t)Mm * Cc) return;
  int row = (int)(i / Cc), c = (int)(i % Cc);
  int b = row / Tt;
  float fac = (pos[(size_t)b * Tt] == 0) ? __expf(-100.0f) : 1.0f;
  float xv = x[i];
  float xxv = shift[(size_t)b * Cc + c] * fac - xv;
  xxxb[i] = f2bf(xv + xxv * maa_x[c]);
}

// ---------------- WMMA bf16 GEMM with async-to-LDS double buffering --------
// C[M,N] = A[M,K](bf16 rowmajor) * BT[N,K](bf16 rowmajor, pre-transposed)
// block tile 128x128, BK=32, 8 waves (wave32), each wave: 2x4 wmma 16x16
#define BM 128
#define BN 128
#define BK 32
#define LDSP 40              /* padded LDS row stride in halfs (80B) */
#define TILEH (BM * LDSP)    /* halfs per A/B tile buffer */

__global__ __launch_bounds__(256) void wmma_gemm_bf16(
    const unsigned short* __restrict__ Amat,
    const unsigned short* __restrict__ BTmat, void* __restrict__ Cout,
    int Mdim, int Ndim, int Kdim, int epi, const float* __restrict__ bias) {
  __shared__ __align__(16) unsigned short lds[2][2][TILEH];
  int tid = threadIdx.x;
  int lane = tid & 31, wave = tid >> 5;
  int wm = wave >> 1, wn = wave & 1;
  int row0 = blockIdx.y * BM;
  int col0 = blockIdx.x * BN;
  int lrow = lane & 15;
  int hsel = lane >> 4;

  // per-thread staging slots: two 16B chunks per matrix per tile
  int sr = tid >> 1;          // 0..127 : tile row
  int scc = (tid & 1) * 16;   // 0 or 16 halfs
  int ga_row = row0 + sr; if (ga_row >= Mdim) ga_row = Mdim - 1; // clamp so
  int gb_row = col0 + sr; if (gb_row >= Ndim) gb_row = Ndim - 1; // EXEC stays full
  const unsigned short* gA = Amat + (size_t)ga_row * Kdim + scc;
  const unsigned short* gB = BTmat + (size_t)gb_row * Kdim + scc;
  unsigned la[2], lb[2];
  la[0] = (unsigned)(size_t)&lds[0][0][sr * LDSP + scc];
  la[1] = (unsigned)(size_t)&lds[1][0][sr * LDSP + scc];
  lb[0] = (unsigned)(size_t)&lds[0][1][sr * LDSP + scc];
  lb[1] = (unsigned)(size_t)&lds[1][1][sr * LDSP + scc];

  v8f zero = {};
  v8f acc[2][4];
#pragma unroll
  for (int i = 0; i < 2; i++)
#pragma unroll
    for (int j = 0; j < 4; j++) acc[i][j] = zero;

  int ntiles = Kdim / BK;
  { // prologue: async-stage tile 0 into buffer 0 (ASYNCcnt += 4)
    asm volatile("global_load_async_to_lds_b128 %0, %1, off" ::"v"(la[0]),
                 "v"(gA)
                 : "memory");
    asm volatile("global_load_async_to_lds_b128 %0, %1, off offset:16" ::"v"(
                     la[0]),
                 "v"(gA)
                 : "memory");
    asm volatile("global_load_async_to_lds_b128 %0, %1, off" ::"v"(lb[0]),
                 "v"(gB)
                 : "memory");
    asm volatile("global_load_async_to_lds_b128 %0, %1, off offset:16" ::"v"(
                     lb[0]),
                 "v"(gB)
                 : "memory");
  }

  for (int it = 0; it < ntiles; ++it) {
    if (it + 1 < ntiles) { // prefetch next tile into the other buffer
      int nb = (it + 1) & 1;
      const unsigned short* pa = gA + (size_t)(it + 1) * BK;
      const unsigned short* pb = gB + (size_t)(it + 1) * BK;
      asm volatile("global_load_async_to_lds_b128 %0, %1, off" ::"v"(la[nb]),
                   "v"(pa)
                   : "memory");
      asm volatile("global_load_async_to_lds_b128 %0, %1, off offset:16" ::"v"(
                       la[nb]),
                   "v"(pa)
                   : "memory");
      asm volatile("global_load_async_to_lds_b128 %0, %1, off" ::"v"(lb[nb]),
                   "v"(pb)
                   : "memory");
      asm volatile("global_load_async_to_lds_b128 %0, %1, off offset:16" ::"v"(
                       lb[nb]),
                   "v"(pb)
                   : "memory");
      // async loads complete in order: <=4 outstanding == current tile landed
      asm volatile("s_wait_asynccnt 0x4" ::: "memory");
    } else {
      asm volatile("s_wait_asynccnt 0x0" ::: "memory");
    }
    __syncthreads();

    const unsigned short* Asb = &lds[it & 1][0][0];
    const unsigned short* Bsb = &lds[it & 1][1][0];

    // A fragment: 16-bit 16x32 layout (half-wave K 0-7/16-23 vs 8-15/24-31)
    v16bf afrag[2], bfrag[4];
    int akb = hsel * 8;
#pragma unroll
    for (int mi = 0; mi < 2; mi++) {
      int r = wm * 32 + mi * 16 + lrow;
      uint4 lo = *(const uint4*)&Asb[r * LDSP + akb];
      uint4 hi = *(const uint4*)&Asb[r * LDSP + akb + 16];
      v16bf a;
      ((uint4*)&a)[0] = lo;
      ((uint4*)&a)[1] = hi;
      afrag[mi] = a;
    }
    // B fragment: 16-bit 32x16 layout (half-wave K 0-7/16-23 vs 8-15/24-31)
    int bkb = hsel * 8;
#pragma unroll
    for (int ni = 0; ni < 4; ni++) {
      int n = wn * 64 + ni * 16 + lrow;
      uint4 lo = *(const uint4*)&Bsb[n * LDSP + bkb];
      uint4 hi = *(const uint4*)&Bsb[n * LDSP + bkb + 16];
      v16bf b;
      ((uint4*)&b)[0] = lo;
      ((uint4*)&b)[1] = hi;
      bfrag[ni] = b;
    }
#pragma unroll
    for (int mi = 0; mi < 2; mi++)
#pragma unroll
      for (int ni = 0; ni < 4; ni++)
        acc[mi][ni] = __builtin_amdgcn_wmma_f32_16x16x32_bf16(
            false, afrag[mi], false, bfrag[ni], (short)0, acc[mi][ni], false,
            false);
    __syncthreads(); // all waves done reading before this buffer is re-staged
  }

  // epilogue: C layout lane->(N=lane&15), vgpr j -> M = j + (lane>>4)*8
  int msel = hsel * 8;
#pragma unroll
  for (int mi = 0; mi < 2; mi++)
#pragma unroll
    for (int ni = 0; ni < 4; ni++) {
      int baser = row0 + wm * 32 + mi * 16 + msel;
      int basec = col0 + wn * 64 + ni * 16 + lrow;
      if (basec < Ndim) {
#pragma unroll
        for (int j = 0; j < 8; j++) {
          int rr = baser + j;
          if (rr < Mdim) {
            size_t oidx = (size_t)rr * Ndim + basec;
            float val = acc[mi][ni][j];
            if (epi == 1)
              ((float*)Cout)[oidx] = tanhf(val);
            else if (epi == 2)
              ((unsigned short*)Cout)[oidx] = f2bf(tanhf(val));
            else if (epi == 3)
              ((float*)Cout)[oidx] = val * (1.0f / (1.0f + __expf(-val)));
            else if (epi == 4)
              ((float*)Cout)[oidx] = -__expf(bias[basec] + val);
            else
              ((float*)Cout)[oidx] = val;
          }
        }
      }
    }
}

// ---------------- mix stage 2: 32-dot + combine, 5 outputs ----------------
__global__ __launch_bounds__(256) void mix_combine_kernel(
    const float* __restrict__ x, const int* __restrict__ pos,
    const float* __restrict__ shift, const float* __restrict__ mixtmp,
    const float* __restrict__ w2t, const float* __restrict__ maa_w,
    const float* __restrict__ maa_k, const float* __restrict__ maa_v,
    const float* __restrict__ maa_r, const float* __restrict__ maa_g,
    unsigned short* __restrict__ outb) {
  int row = blockIdx.x;
  __shared__ float mrow[5 * TMX];
  if (threadIdx.x < 5 * TMX)
    mrow[threadIdx.x] = mixtmp[(size_t)row * (5 * TMX) + threadIdx.x];
  __syncthreads();
  int b = row / Tt;
  float fac = (pos[(size_t)b * Tt] == 0) ? __expf(-100.0f) : 1.0f;
  const float* maas[5] = {maa_w, maa_k, maa_v, maa_r, maa_g};
  for (int c = threadIdx.x; c < Cc; c += 256) {
    float xv = x[(size_t)row * Cc + c];
    float xxv = shift[(size_t)b * Cc + c] * fac - xv;
#pragma unroll
    for (int f = 0; f < 5; f++) {
      const float* wv = w2t + ((size_t)f * Cc + c) * TMX;
      const float* mt = mrow + f * TMX;
      float dot = 0.0f;
#pragma unroll
      for (int e = 0; e < TMX; e++) dot += mt[e] * wv[e];
      float val = xv + xxv * (maas[f][c] + dot);
      outb[((size_t)f * Mm + row) * Cc + c] = f2bf(val);
    }
  }
}

// ---------------- sequential RWKV scan: one block per (b,h) ----------------
__global__ __launch_bounds__(64) void rwkv_scan_kernel(
    const float* __restrict__ r, const float* __restrict__ k,
    const float* __restrict__ v, const float* __restrict__ w,
    const float* __restrict__ kv0, const float* __restrict__ u,
    const int* __restrict__ pos, float* __restrict__ o) {
  int bh = blockIdx.x;
  int b = bh / Hh, h = bh - b * Hh;
  int vc = threadIdx.x; // value-dim column owned by this thread
  float S[Zz];
  const float* s0 = kv0 + (size_t)bh * Zz * Zz;
#pragma unroll
  for (int kk = 0; kk < Zz; kk++) S[kk] = s0[kk * Zz + vc];
  __shared__ float4 rkw[Zz];
  __shared__ float uu[Zz];
  uu[vc] = u[h * Zz + vc];
  float reset = (pos[(size_t)b * Tt] == 0) ? -100.0f : 0.0f;
  size_t base = (size_t)b * Tt * Aa + (size_t)h * Zz + vc;
  for (int t = 0; t < Tt; t++) {
    size_t idx = base + (size_t)t * Aa;
    float rv = r[idx], kvl = k[idx], wv = w[idx], vvv = v[idx];
    if (t == 0) wv += reset;
    __syncthreads();
    rkw[vc] = make_float4(rv, kvl, __expf(wv), 0.0f);
    __syncthreads();
    float oacc = 0.0f;
#pragma unroll
    for (int kk = 0; kk < Zz; kk++) {
      float4 q = rkw[kk]; // broadcast ds_load_b128
      float kvp = q.y * vvv;
      oacc += q.x * (S[kk] + uu[kk] * kvp);
      S[kk] = q.z * S[kk] + kvp;
    }
    o[idx] = oacc;
  }
}

// ---------------- GroupNorm(H groups) * silu(g) -> bf16 ----------------
__global__ __launch_bounds__(256) void gnorm_gate_kernel(
    const float* __restrict__ o, const float* __restrict__ g,
    const float* __restrict__ ln_w, const float* __restrict__ ln_b,
    unsigned short* __restrict__ gatedb) {
  int row = blockIdx.x;
  int tid = threadIdx.x;
  int c0 = tid * 8; // 8 cols per thread; 8 threads per head (64 cols)
  size_t base = (size_t)row * Aa + c0;
  float vals[8];
  float s = 0.f, s2 = 0.f;
#pragma unroll
  for (int i = 0; i < 8; i++) {
    float xv = o[base + i];
    vals[i] = xv;
    s += xv;
    s2 += xv * xv;
  }
#pragma unroll
  for (int m = 1; m < 8; m <<= 1) { // reduce within aligned 8-lane head group
    s += __shfl_xor(s, m, 32);
    s2 += __shfl_xor(s2, m, 32);
  }
  float mean = s * (1.0f / Zz);
  float var = s2 * (1.0f / Zz) - mean * mean;
  float rs = rsqrtf(var + GN_EPS);
#pragma unroll
  for (int i = 0; i < 8; i++) {
    float xn = (vals[i] - mean) * rs * ln_w[c0 + i] + ln_b[c0 + i];
    gatedb[base + i] = f2bf(xn * g[base + i]);
  }
}

// ---------------- launch ----------------
extern "C" void kernel_launch(void* const* d_in, const int* in_sizes, int n_in,
                              void* d_out, int out_size, void* d_ws,
                              size_t ws_size, hipStream_t stream) {
  (void)in_sizes;
  (void)n_in;
  (void)out_size;
  (void)ws_size;
  const float* x = (const float*)d_in[0];
  const int* pos = (const int*)d_in[1];
  const float* kv0 = (const float*)d_in[2];
  const float* shift = (const float*)d_in[3];
  const float* maa_x = (const float*)d_in[4];
  const float* maa_w = (const float*)d_in[5];
  const float* maa_k = (const float*)d_in[6];
  const float* maa_v = (const float*)d_in[7];
  const float* maa_r = (const float*)d_in[8];
  const float* maa_g = (const float*)d_in[9];
  const float* w1 = (const float*)d_in[10];
  const float* w2 = (const float*)d_in[11];
  const float* tdecay = (const float*)d_in[12];
  const float* dw1 = (const float*)d_in[13];
  const float* dw2 = (const float*)d_in[14];
  const float* faaaa = (const float*)d_in[15];
  const float* W_r = (const float*)d_in[16];
  const float* W_k = (const float*)d_in[17];
  const float* W_v = (const float*)d_in[18];
  const float* W_g = (const float*)d_in[19];
  const float* W_o = (const float*)d_in[20];
  const float* ln_w = (const float*)d_in[21];
  const float* ln_b = (const float*)d_in[22];

  char* ws = (char*)d_ws;
  size_t off = 0;
  auto alloc = [&](size_t bytes) -> void* {
    void* p = ws + off;
    off += (bytes + 255) & ~(size_t)255;
    return p;
  };
  unsigned short* xxxb = (unsigned short*)alloc((size_t)Mm * Cc * 2);
  float* mixtmp = (float*)alloc((size_t)Mm * 5 * TMX * 4);
  unsigned short* mixedb = (unsigned short*)alloc((size_t)5 * Mm * Cc * 2);
  unsigned short* w1T = (unsigned short*)alloc((size_t)Cc * 5 * TMX * 2);
  float* w2t = (float*)alloc((size_t)5 * Cc * TMX * 4);
  unsigned short* dw1T = (unsigned short*)alloc((size_t)Cc * TDX * 2);
  unsigned short* dw2T = (unsigned short*)alloc((size_t)TDX * Aa * 2);
  unsigned short* WrT = (unsigned short*)alloc((size_t)Cc * Aa * 2);
  unsigned short* WkT = (unsigned short*)alloc((size_t)Cc * Aa * 2);
  unsigned short* WvT = (unsigned short*)alloc((size_t)Cc * Aa * 2);
  unsigned short* WgT = (unsigned short*)alloc((size_t)Cc * Aa * 2);
  unsigned short* WoT = (unsigned short*)alloc((size_t)Aa * Cc * 2);
  float* rbuf = (float*)alloc((size_t)Mm * Aa * 4);
  float* kbuf = (float*)alloc((size_t)Mm * Aa * 4);
  float* vbuf = (float*)alloc((size_t)Mm * Aa * 4);
  float* wbuf = (float*)alloc((size_t)Mm * Aa * 4);
  float* gbuf = (float*)alloc((size_t)Mm * Aa * 4);
  unsigned short* tdtmpb = (unsigned short*)alloc((size_t)Mm * TDX * 2);
  float* obuf = (float*)alloc((size_t)Mm * Aa * 4);
  unsigned short* gatedb = (unsigned short*)alloc((size_t)Mm * Aa * 2);

  dim3 blk(256);
  // pre-transposed bf16 weights: out[N,K] from in[K,N]
  auto cvtT = [&](const float* src, unsigned short* dst, int K, int N) {
    size_t n = (size_t)K * N;
    cvt_bf16_t_kernel<<<dim3((unsigned)((n + 255) / 256)), blk, 0, stream>>>(
        src, dst, K, N);
  };
  cvtT(W_r, WrT, Cc, Aa);
  cvtT(W_k, WkT, Cc, Aa);
  cvtT(W_v, WvT, Cc, Aa);
  cvtT(W_g, WgT, Cc, Aa);
  cvtT(W_o, WoT, Aa, Cc);
  cvtT(w1, w1T, Cc, 5 * TMX);
  cvtT(dw1, dw1T, Cc, TDX);
  cvtT(dw2, dw2T, TDX, Aa);
  w2_transpose_kernel<<<dim3((5 * TMX * Cc + 255) / 256), blk, 0, stream>>>(
      w2, w2t);
  prep_xxx_kernel<<<dim3((Mm * Cc + 255) / 256), blk, 0, stream>>>(
      x, pos, shift, maa_x, xxxb);

  auto gemm = [&](const unsigned short* A_, const unsigned short* BT_, void* C_,
                  int Md, int Nd, int Kd, int epi, const float* bias) {
    dim3 grid((Nd + BN - 1) / BN, (Md + BM - 1) / BM);
    wmma_gemm_bf16<<<grid, blk, 0, stream>>>(A_, BT_, C_, Md, Nd, Kd, epi,
                                             bias);
  };

  // stage 1: mixtmp = tanh(xxx @ w1)   [M,160]
  gemm(xxxb, w1T, mixtmp, Mm, 5 * TMX, Cc, /*tanh*/ 1, nullptr);
  // stage 2: five mixed bf16 inputs (tdin, kx, vx, rx, gx)
  mix_combine_kernel<<<dim3(Mm), blk, 0, stream>>>(
      x, pos, shift, mixtmp, w2t, maa_w, maa_k, maa_v, maa_r, maa_g, mixedb);

  const unsigned short* tdin = mixedb + (size_t)0 * Mm * Cc;
  const unsigned short* kx = mixedb + (size_t)1 * Mm * Cc;
  const unsigned short* vx = mixedb + (size_t)2 * Mm * Cc;
  const unsigned short* rx = mixedb + (size_t)3 * Mm * Cc;
  const unsigned short* gx = mixedb + (size_t)4 * Mm * Cc;

  // projections
  gemm(rx, WrT, rbuf, Mm, Aa, Cc, 0, nullptr);
  gemm(kx, WkT, kbuf, Mm, Aa, Cc, 0, nullptr);
  gemm(vx, WvT, vbuf, Mm, Aa, Cc, 0, nullptr);
  gemm(gx, WgT, gbuf, Mm, Aa, Cc, /*silu*/ 3, nullptr);
  // decay: tdtmp = tanh(tdin @ dw1) (bf16), wlog = -exp(tdecay + tdtmp @ dw2)
  gemm(tdin, dw1T, tdtmpb, Mm, TDX, Cc, /*tanh->bf16*/ 2, nullptr);
  gemm(tdtmpb, dw2T, wbuf, Mm, Aa, TDX, /*-exp(bias+x)*/ 4, tdecay);

  // sequential scan over T per (b,h)
  rwkv_scan_kernel<<<dim3(Bb * Hh), dim3(64), 0, stream>>>(
      rbuf, kbuf, vbuf, wbuf, kv0, faaaa, pos, obuf);

  // GroupNorm + gate -> bf16
  gnorm_gate_kernel<<<dim3(Mm), blk, 0, stream>>>(obuf, gbuf, ln_w, ln_b,
                                                  gatedb);
  // out = gated @ W_o  [M, C] f32
  gemm(gatedb, WoT, (float*)d_out, Mm, Cc, Aa, 0, nullptr);
}
